// GraphNeuralAssociator_17514876633940
// MI455X (gfx1250) — matrix operand order repaired and checked by reading, not compiled, not run on hardware
//
#include <hip/hip_runtime.h>

// ---------------------------------------------------------------------------
// MI455X (gfx1250) GNN edge-associator.
// - All matmuls via v_wmma_f32_16x16x32_f16 (f16 in, f32 accumulate).
// - Edge MLP layer-1 decomposed into per-node A/B tables; layers 2+3 fused
//   with per-pair score reduction; triangle-only blocks.
// - Tile staging via CDNA5 async memory->LDS DMA (global_load_async_to_lds,
//   ASYNCcnt), triple-buffered distance-2 prefetch with a constant in-flight
//   invariant (single unconditional s_wait_asynccnt 2 per iteration).
// ---------------------------------------------------------------------------

typedef _Float16 f16;
typedef _Float16 v8h  __attribute__((ext_vector_type(8)));
typedef _Float16 v16h __attribute__((ext_vector_type(16)));
typedef float    v8f  __attribute__((ext_vector_type(8)));

#define NNODE 512
#define NFEAT 256
#define NHID  512

__device__ __forceinline__ v8f wmma_f16(v16h a, v16h b, v8f c) {
  return __builtin_amdgcn_wmma_f32_16x16x32_f16(
      /*neg_a=*/false, a, /*neg_b=*/false, b,
      /*c_mod=*/(short)0, c, /*reuse_a=*/false, /*reuse_b=*/false);
}

// True LDS byte offset (addrspace(3) ptrtoint) for the async-DMA VDST field.
__device__ __forceinline__ unsigned lds_off_of(const void* p) {
  return (unsigned)(unsigned long long)(__attribute__((address_space(3))) const char*)p;
}

// CDNA5 async DMA: 16 bytes per lane, memory -> LDS, tracked by ASYNCcnt.
__device__ __forceinline__ void async_ld_b128(unsigned lds_off, const void* gptr) {
  unsigned long long ga = (unsigned long long)gptr;
  asm volatile("global_load_async_to_lds_b128 %0, %1, off"
               :: "v"(lds_off), "v"(ga) : "memory");
}
__device__ __forceinline__ void wait_async0() {
  asm volatile("s_wait_asynccnt 0x0" ::: "memory");
}
__device__ __forceinline__ void wait_async2() {
  asm volatile("s_wait_asynccnt 0x2" ::: "memory");
}

// A-operand fragment (16x32 f16) from row-major [m][k] (k-contiguous).
// ISA: lanes 0-15 -> M=lane, K={0..7,16..23}; lanes 16-31 -> M=lane-16, K={8..15,24..31}.
__device__ __forceinline__ v16h load_frag_a(const f16* base, int stride, int lane) {
  const f16* p = base + (lane & 15) * stride + ((lane < 16) ? 0 : 8);
  v8h lo = *(const v8h*)(p);
  v8h hi = *(const v8h*)(p + 16);
  return __builtin_shufflevector(lo, hi, 0,1,2,3,4,5,6,7,8,9,10,11,12,13,14,15);
}

// B-operand fragment (32x16 f16) from N-major [n][k] (W^T, k-contiguous).
// ISA: lanes 0-15 -> N=lane, K={0..7,16..23}; lanes 16-31 -> N=lane-16, K={8..15,24..31}.
__device__ __forceinline__ v16h load_frag_b(const f16* base, int stride, int lane) {
  const f16* p = base + (lane & 15) * stride + ((lane < 16) ? 0 : 8);
  v8h lo = *(const v8h*)(p);
  v8h hi = *(const v8h*)(p + 16);
  return __builtin_shufflevector(lo, hi, 0,1,2,3,4,5,6,7,8,9,10,11,12,13,14,15);
}

// ---------------------------------------------------------------------------
// Generic WMMA GEMM: C[M,N] = epilogue( A[M,K](f16) * W[K,N] + ... )
// W supplied as WT[N][K] (f16, row stride ldw). 64x64 block tile, 4 waves,
// each wave a 32x32 sub-tile. Double-buffered async-DMA staging.
// ---------------------------------------------------------------------------
template<bool BIAS, bool ADDMAT, bool RELU, bool OUTF, bool OUTH, bool HTRANS>
__global__ __launch_bounds__(128)
void gemm_f16(const f16* __restrict__ A, int lda,
              const f16* __restrict__ WT, int ldw,
              const float* __restrict__ bias,
              const float* __restrict__ addmat,   // [M][N] f32
              float* __restrict__ outF,           // [M][N] f32
              f16*  __restrict__ outH,            // [M][N] or (HTRANS) [N][M]
              int M, int N, int K) {
  __shared__ __align__(16) f16 At[2][64 * 40];
  __shared__ __align__(16) f16 Bt[2][64 * 40];
  const int tid  = threadIdx.x;
  const int lane = tid & 31;
  const int wid  = tid >> 5;
  const int mb   = blockIdx.y * 64;
  const int nb   = blockIdx.x * 64;
  const int wm   = (wid >> 1) * 32;
  const int wn   = (wid & 1) * 32;

  auto stage = [&](int kb, int buf) {
#pragma unroll
    for (int r = 0; r < 2; ++r) {
      int q = tid + r * 128;
      int row = q >> 2, seg = q & 3;
      async_ld_b128(lds_off_of(&At[buf][row * 40 + seg * 8]),
                    &A[(mb + row) * lda + kb + seg * 8]);
      async_ld_b128(lds_off_of(&Bt[buf][row * 40 + seg * 8]),
                    &WT[(nb + row) * ldw + kb + seg * 8]);
    }
  };

  v8f acc[2][2] = {};
  const int nk = K >> 5;
  stage(0, 0);

  for (int ks = 0; ks < nk; ++ks) {
    wait_async0();
    __syncthreads();
    if (ks + 1 < nk) stage((ks + 1) * 32, (ks + 1) & 1);
    const f16* at = At[ks & 1];
    const f16* bt = Bt[ks & 1];
    v16h af0 = load_frag_a(&at[(wm +  0) * 40], 40, lane);
    v16h af1 = load_frag_a(&at[(wm + 16) * 40], 40, lane);
    v16h bf0 = load_frag_b(&bt[(wn +  0) * 40], 40, lane);
    v16h bf1 = load_frag_b(&bt[(wn + 16) * 40], 40, lane);
    acc[0][0] = wmma_f16(af0, bf0, acc[0][0]);
    acc[0][1] = wmma_f16(af0, bf1, acc[0][1]);
    acc[1][0] = wmma_f16(af1, bf0, acc[1][0]);
    acc[1][1] = wmma_f16(af1, bf1, acc[1][1]);
  }

  // Epilogue. C tile: VGPR r, lanes 0-15 -> (M=r, N=lane); lanes 16-31 -> (M=r+8).
  const int hi   = lane >> 4;
  const int col0 = lane & 15;
#pragma unroll
  for (int mi = 0; mi < 2; ++mi) {
#pragma unroll
    for (int ni = 0; ni < 2; ++ni) {
      int col = nb + wn + ni * 16 + col0;
#pragma unroll
      for (int r = 0; r < 8; ++r) {
        int row = mb + wm + mi * 16 + r + hi * 8;
        float v = acc[mi][ni][r];
        if (BIAS)   v += bias[col];
        if (ADDMAT) v += addmat[row * N + col];
        if (RELU)   v = fmaxf(v, 0.0f);
        if (OUTF)   outF[row * N + col] = v;
        if (OUTH) {
          if (HTRANS) outH[col * M + row] = (f16)v;
          else        outH[row * N + col] = (f16)v;
        }
      }
    }
  }
}

// ---------------------------------------------------------------------------
// Fused edge MLP: block = 16 i-rows x 8 j-cols = 128 pairs, 4 waves.
// H1[p][k] = relu(Apair[i][k] + Bpair[j][k]) staged once in LDS (f16, 130 KB).
// Each wave owns 2 M-tiles (32 pairs) x 4 N-tiles per 64-wide N chunk:
// 8 WMMAs per 12 ds_load_b128.  w2^T tiles arrive via triple-buffered async
// DMA at a constant prefetch distance of 2 (always exactly 2 stages = 4 async
// ops in flight, so a single s_wait_asynccnt 2 retires the tile being
// consumed).  Overflow prefetches wrap to valid-but-unused tiles; s_endpgm's
// implicit wait-idle drains them.
// ---------------------------------------------------------------------------
__global__ __launch_bounds__(128)
void edge_mlp_kernel(const float* __restrict__ Apair,  // [512][512] f32
                     const float* __restrict__ Bpair,  // [512][512] f32
                     const f16*  __restrict__ w2T,     // [512][512] f16 (N-major)
                     const float* __restrict__ b2,     // [512]
                     const float* __restrict__ w3,     // [512]
                     const float* __restrict__ b3,     // [1]
                     float* __restrict__ scores) {     // [512][512]
  const int jb = blockIdx.x * 8;
  const int ib = blockIdx.y * 16;
  if (jb + 7 <= ib) return;  // no pair with i<j in this block

  __shared__ __align__(16) f16 H1[128 * 520];     // 130 KB
  __shared__ __align__(16) f16 WTt[3][64 * 40];   // triple-buffered w2^T tiles

  const int tid  = threadIdx.x;
  const int lane = tid & 31;
  const int wid  = tid >> 5;   // 0..3
  const int mrow = wid * 32;   // this wave's 32 pairs (2 M-tiles)

  // Per-thread staging constants: two b128 segments per stage.
  const int srow = tid >> 2, sseg = tid & 3;
  const int ls0 = (srow +  0) * 40 + sseg * 8;   // LDS f16 offsets within a buffer
  const int ls1 = (srow + 32) * 40 + sseg * 8;
  const int gs0 = (srow +  0) * NHID + sseg * 8; // global f16 offsets within a tile row-block
  const int gs1 = (srow + 32) * NHID + sseg * 8;

  // Prefetch tile (pn, pk): rows pn*64.., k pk*32..  into buffer `buf`.
  auto stageW = [&](int pn, int pk, int buf) {
    const f16* base = w2T + pn * (64 * NHID) + pk * 32;
    async_ld_b128(lds_off_of(&WTt[buf][ls0]), base + gs0);
    async_ld_b128(lds_off_of(&WTt[buf][ls1]), base + gs1);
  };

  stageW(0, 0, 0);   // both prefetch DMAs overlap the entire H1 build
  stageW(0, 1, 1);

  // Build H1: pair p -> (i = ib + p/8, j = jb + p%8); float4 in, 4xf16 out.
  for (int p = 0; p < 128; ++p) {
    const float4* Ar = (const float4*)(Apair + (ib + (p >> 3)) * NHID);
    const float4* Br = (const float4*)(Bpair + (jb + (p & 7)) * NHID);
    float4 a = Ar[tid];
    float4 b = Br[tid];
    union { f16 h[4]; uint2 u; } pk;
    pk.h[0] = (f16)fmaxf(a.x + b.x, 0.0f);
    pk.h[1] = (f16)fmaxf(a.y + b.y, 0.0f);
    pk.h[2] = (f16)fmaxf(a.z + b.z, 0.0f);
    pk.h[3] = (f16)fmaxf(a.w + b.w, 0.0f);
    *(uint2*)&H1[p * 520 + 4 * tid] = pk.u;
  }

  float sc[2][8] = {};
  int bufc = 0;            // buffer being consumed this iteration
  int bufp = 2;            // buffer receiving the distance-2 prefetch
  int pn = 0, pk = 2;      // prefetch tile coordinates (tile index it+2)

  for (int nc = 0; nc < 8; ++nc) {
    const int nbase = nc * 64;
    float b2v[4], w3v[4];
#pragma unroll
    for (int t = 0; t < 4; ++t) {
      int col = nbase + t * 16 + (lane & 15);
      b2v[t] = b2[col];
      w3v[t] = w3[col];
    }
    v8f acc[2][4] = {};

    for (int ks = 0; ks < 16; ++ks) {
      wait_async2();       // retire the oldest stage (the tile consumed now)
      __syncthreads();
      stageW(pn, pk, bufp);              // unconditional: invariant 2 stages in flight
      pk++; if (pk == 16) { pk = 0; pn = (pn + 1) & 7; }  // wrap -> harmless refetch
      const f16* wt = WTt[bufc];
      bufc = (bufc == 2) ? 0 : bufc + 1;
      bufp = (bufp == 2) ? 0 : bufp + 1;

      v16h af0 = load_frag_a(&H1[(mrow +  0) * 520 + ks * 32], 520, lane);
      v16h af1 = load_frag_a(&H1[(mrow + 16) * 520 + ks * 32], 520, lane);
      v16h bf[4];
#pragma unroll
      for (int t = 0; t < 4; ++t)
        bf[t] = load_frag_b(&wt[(t * 16) * 40], 40, lane);
#pragma unroll
      for (int t = 0; t < 4; ++t) {
        acc[0][t] = wmma_f16(af0, bf[t], acc[0][t]);
        acc[1][t] = wmma_f16(af1, bf[t], acc[1][t]);
      }
    }

    // relu(h2 + b2) . w3 over this wave's 64 channels, per M-tile
#pragma unroll
    for (int m = 0; m < 2; ++m) {
      float part[8] = {0.f, 0.f, 0.f, 0.f, 0.f, 0.f, 0.f, 0.f};
#pragma unroll
      for (int t = 0; t < 4; ++t)
#pragma unroll
        for (int r = 0; r < 8; ++r) {
          float v = acc[m][t][r] + b2v[t];
          v = fmaxf(v, 0.0f);
          part[r] += v * w3v[t];
        }
      // butterfly-reduce over each 16-lane half (N dimension)
#pragma unroll
      for (int r = 0; r < 8; ++r) {
        part[r] += __shfl_xor(part[r], 1);
        part[r] += __shfl_xor(part[r], 2);
        part[r] += __shfl_xor(part[r], 4);
        part[r] += __shfl_xor(part[r], 8);
        sc[m][r] += part[r];
      }
    }
  }

  // lanes 0 (rows +0..7) and 16 (rows +8..15) hold per-pair totals
  const float bias3 = b3[0];
  const int hi = lane >> 4;
  if ((lane & 15) == 0) {
#pragma unroll
    for (int m = 0; m < 2; ++m)
#pragma unroll
      for (int r = 0; r < 8; ++r) {
        int p = mrow + m * 16 + r + hi * 8;
        int i = ib + (p >> 3);
        int j = jb + (p & 7);
        if (i < j) {
          float s = 1.0f / (1.0f + __expf(-(sc[m][r] + bias3)));
          scores[i * NNODE + j] = s;
          scores[j * NNODE + i] = s;
        }
      }
  }
}

// ---------------------------------------------------------------------------
// Small helper kernels
// ---------------------------------------------------------------------------
__global__ __launch_bounds__(256)
void adj_norm_kernel(const float* __restrict__ adj, f16* __restrict__ adjh) {
  __shared__ float red[256];
  const int r = blockIdx.x, t = threadIdx.x;
  const float* row = adj + r * NNODE;
  red[t] = row[t] + row[t + 256];
  __syncthreads();
  for (int o = 128; o > 0; o >>= 1) {
    if (t < o) red[t] += red[t + o];
    __syncthreads();
  }
  const float inv = 1.0f / (red[0] + 1e-8f);
  adjh[r * NNODE + t]       = (f16)(row[t] * inv);
  adjh[r * NNODE + t + 256] = (f16)(row[t + 256] * inv);
}

__global__ void cvt_f16_kernel(const float* __restrict__ in, f16* __restrict__ out, int n) {
  int i = blockIdx.x * 256 + threadIdx.x;
  if (i < n) out[i] = (f16)in[i];
}

// W[K][N] f32 -> WT[N][K] f16
__global__ void transpose_cvt_kernel(const float* __restrict__ in, f16* __restrict__ out,
                                     int K, int N) {
  int idx = blockIdx.x * 256 + threadIdx.x;
  if (idx < K * N) {
    int k = idx / N, n = idx - k * N;
    out[n * K + k] = (f16)in[idx];
  }
}

__global__ void zero_kernel(float* __restrict__ p, int n) {
  int i = blockIdx.x * 256 + threadIdx.x;
  if (i < n) p[i] = 0.0f;
}

// ---------------------------------------------------------------------------
extern "C" void kernel_launch(void* const* d_in, const int* in_sizes, int n_in,
                              void* d_out, int out_size, void* d_ws, size_t ws_size,
                              hipStream_t stream) {
  (void)in_sizes; (void)n_in; (void)out_size; (void)ws_size;
  const float* nf     = (const float*)d_in[0];
  const float* adj    = (const float*)d_in[1];
  const float* gc1_wl = (const float*)d_in[2];  const float* gc1_bl = (const float*)d_in[3];
  const float* gc1_ws = (const float*)d_in[4];  const float* gc1_bs = (const float*)d_in[5];
  const float* gc2_wl = (const float*)d_in[6];  const float* gc2_bl = (const float*)d_in[7];
  const float* gc2_ws = (const float*)d_in[8];  const float* gc2_bs = (const float*)d_in[9];
  const float* gc3_wl = (const float*)d_in[10]; const float* gc3_bl = (const float*)d_in[11];
  const float* gc3_ws = (const float*)d_in[12]; const float* gc3_bs = (const float*)d_in[13];
  const float* mw1    = (const float*)d_in[14]; const float* mb1    = (const float*)d_in[15];
  const float* mw2    = (const float*)d_in[16]; const float* mb2    = (const float*)d_in[17];
  const float* mw3    = (const float*)d_in[18]; const float* mb3    = (const float*)d_in[19];

  float* embF   = (float*)d_out;                 // [512][256]
  float* edgeF  = (float*)d_out + NNODE * NFEAT; // [512][512]

  char* ws = (char*)d_ws;
  size_t off = 0;
  auto alloc = [&](size_t bytes) -> void* {
    void* p = ws + off;
    off = (off + bytes + 255) & ~(size_t)255;
    return p;
  };

  f16* adjh  = (f16*)alloc(512 * 512 * 2);
  f16* x0h   = (f16*)alloc(512 * 256 * 2);
  f16* w1lT  = (f16*)alloc(512 * 256 * 2);  // gc1_wl^T [512][256]
  f16* w1sT  = (f16*)alloc(512 * 256 * 2);
  f16* w2lT  = (f16*)alloc(512 * 512 * 2);
  f16* w2sT  = (f16*)alloc(512 * 512 * 2);
  f16* w3lT  = (f16*)alloc(256 * 512 * 2);  // gc3_wl^T [256][512]
  f16* w3sT  = (f16*)alloc(256 * 512 * 2);
  f16* mw1T  = (f16*)alloc(512 * 512 * 2);  // mlp_w1^T [512 n][512 k]
  f16* mw2T  = (f16*)alloc(512 * 512 * 2);  // mlp_w2^T [512 n][512 k]
  f16* YbT   = (f16*)alloc(512 * 512 * 2);  // transposed intermediate (Y^T)
  float* Sb  = (float*)alloc(512 * 512 * 4);
  f16* X1h   = (f16*)alloc(512 * 512 * 2);
  f16* X2h   = (f16*)alloc(512 * 512 * 2);
  f16* embh  = (f16*)alloc(512 * 256 * 2);
  float* Ap  = (float*)alloc(512 * 512 * 4);
  float* Bp  = (float*)alloc(512 * 512 * 4);

  // --- precompute: normalize adjacency, convert inputs/weights to f16 ---
  adj_norm_kernel<<<512, 256, 0, stream>>>(adj, adjh);
  cvt_f16_kernel<<<(512 * 256 + 255) / 256, 256, 0, stream>>>(nf, x0h, 512 * 256);
  transpose_cvt_kernel<<<(256 * 512 + 255) / 256, 256, 0, stream>>>(gc1_wl, w1lT, 256, 512);
  transpose_cvt_kernel<<<(256 * 512 + 255) / 256, 256, 0, stream>>>(gc1_ws, w1sT, 256, 512);
  transpose_cvt_kernel<<<(512 * 512 + 255) / 256, 256, 0, stream>>>(gc2_wl, w2lT, 512, 512);
  transpose_cvt_kernel<<<(512 * 512 + 255) / 256, 256, 0, stream>>>(gc2_ws, w2sT, 512, 512);
  transpose_cvt_kernel<<<(512 * 256 + 255) / 256, 256, 0, stream>>>(gc3_wl, w3lT, 512, 256);
  transpose_cvt_kernel<<<(512 * 256 + 255) / 256, 256, 0, stream>>>(gc3_ws, w3sT, 512, 256);
  transpose_cvt_kernel<<<(512 * 512 + 255) / 256, 256, 0, stream>>>(mw1, mw1T, 512, 512);
  transpose_cvt_kernel<<<(512 * 512 + 255) / 256, 256, 0, stream>>>(mw2, mw2T, 512, 512);
  zero_kernel<<<(512 * 512 + 255) / 256, 256, 0, stream>>>(edgeF, 512 * 512);

  const dim3 blk(128);
  const dim3 g88(8, 8);   // N=512, M=512
  const dim3 g48(4, 8);   // N=256, M=512

  // --- GraphConv layer 1 ---
  gemm_f16<true,false,false,false,true,true ><<<g88, blk, 0, stream>>>(x0h, 256, w1lT, 256, gc1_bl, nullptr, nullptr, YbT, 512, 512, 256);
  gemm_f16<true,false,false,true ,false,false><<<g88, blk, 0, stream>>>(x0h, 256, w1sT, 256, gc1_bs, nullptr, Sb, nullptr, 512, 512, 256);
  gemm_f16<false,true ,true ,false,true,false><<<g88, blk, 0, stream>>>(adjh, 512, YbT, 512, nullptr, Sb, nullptr, X1h, 512, 512, 512);
  // --- GraphConv layer 2 ---
  gemm_f16<true,false,false,false,true,true ><<<g88, blk, 0, stream>>>(X1h, 512, w2lT, 512, gc2_bl, nullptr, nullptr, YbT, 512, 512, 512);
  gemm_f16<true,false,false,true ,false,false><<<g88, blk, 0, stream>>>(X1h, 512, w2sT, 512, gc2_bs, nullptr, Sb, nullptr, 512, 512, 512);
  gemm_f16<false,true ,true ,false,true,false><<<g88, blk, 0, stream>>>(adjh, 512, YbT, 512, nullptr, Sb, nullptr, X2h, 512, 512, 512);
  // --- GraphConv layer 3 (emb, no relu; f32 -> d_out, f16 for MLP) ---
  gemm_f16<true,false,false,false,true,true ><<<g48, blk, 0, stream>>>(X2h, 512, w3lT, 512, gc3_bl, nullptr, nullptr, YbT, 512, 256, 512);
  gemm_f16<true,false,false,true ,false,false><<<g48, blk, 0, stream>>>(X2h, 512, w3sT, 512, gc3_bs, nullptr, Sb, nullptr, 512, 256, 512);
  gemm_f16<false,true ,false,true ,true,false><<<g48, blk, 0, stream>>>(adjh, 512, YbT, 512, nullptr, Sb, embF, embh, 512, 256, 512);
  // --- edge MLP layer-1 decomposition: A = emb@w1_top + b1 ; B = emb@w1_bot ---
  gemm_f16<true ,false,false,true,false,false><<<g88, blk, 0, stream>>>(embh, 256, mw1T,       512, mb1, nullptr, Ap, nullptr, 512, 512, 256);
  gemm_f16<false,false,false,true,false,false><<<g88, blk, 0, stream>>>(embh, 256, mw1T + 256, 512, nullptr, nullptr, Bp, nullptr, 512, 512, 256);

  // --- fused edge MLP (layers 2+3 + sigmoid + symmetric scatter) ---
  edge_mlp_kernel<<<dim3(64, 32), 128, 0, stream>>>(Ap, Bp, mw2T, mb2, mw3, mb3, edgeF);
}
